// TransformerBlock_60344290509345
// MI455X (gfx1250) — matrix operand order repaired and checked by reading, not compiled, not run on hardware
//
#include <hip/hip_runtime.h>
#include <hip/hip_bf16.h>

typedef __attribute__((ext_vector_type(16))) _Float16 v16h;
typedef __attribute__((ext_vector_type(8)))  _Float16 v8h;
typedef __attribute__((ext_vector_type(8)))  float    v8f;

#define BDIM 2
#define NPTS 8192
#define DP   128
#define DMODEL 512
#define KNN  16

// ---------------------------------------------------------------------------
// A-fragment loader (16-bit A, 16x32 MxK, row-major), contiguous k order (same as repack_b_kernel).
// lanes 0-15: M=lane, K=0..15; lanes16-31: M=lane-16, K=16..31
// Each lane's data = two contiguous 16B chunks -> two b128 loads.
__device__ __forceinline__ v16h load_a_frag(const _Float16* A, int lda, int lane) {
  int m   = lane & 15;
  int hlf = (lane >> 4) & 1;
  const _Float16* row = A + (size_t)m * lda + 16 * hlf;
  v8h lo = *(const v8h*)row;          // K = 16*hlf .. +7
  v8h hi = *(const v8h*)(row + 8);    // K = 16*hlf+8 .. +7
  return __builtin_shufflevector(lo, hi, 0, 1, 2, 3, 4, 5, 6, 7,
                                 8, 9, 10, 11, 12, 13, 14, 15);
}

// ---------------------------------------------------------------------------
// Repack (and f32->f16 convert) a weight matrix B[K,N] (row-major) into
// WMMA-fragment-major tiles: P[(kt*(N/16)+nt)*32 + lane] = lane's v16h.
// B-fragment layout (32x16 KxN): lanes 0-15: col=n, K=0..15 (2/VGPR);
//                                lanes16-31: col=n, K=16..31.
__global__ void repack_b_kernel(const float* __restrict__ B,
                                v16h* __restrict__ P, int K, int N) {
  int t = blockIdx.x * blockDim.x + threadIdx.x;
  int total = (K / 32) * (N / 16) * 32;
  if (t >= total) return;
  int lane   = t & 31;
  int tile   = t >> 5;
  int ntiles = N >> 4;
  int kt = tile / ntiles, nt = tile % ntiles;
  int n   = nt * 16 + (lane & 15);
  int hlf = (lane >> 4) & 1;
  v16h v;
#pragma unroll
  for (int g = 0; g < 8; ++g) {
    int k = kt * 32 + 16 * hlf + 2 * g;
    v[2 * g]     = (_Float16)B[(size_t)k * N + n];
    v[2 * g + 1] = (_Float16)B[(size_t)(k + 1) * N + n];
  }
  P[t] = v;
}

// ---------------------------------------------------------------------------
// f32 -> f16 convert (activations)
__global__ void convert_f16_kernel(const float* __restrict__ in,
                                   _Float16* __restrict__ out, int n) {
  int i = blockIdx.x * blockDim.x + threadIdx.x;
  if (i < n) out[i] = (_Float16)in[i];
}

// ---------------------------------------------------------------------------
// Brute-force KNN: one wave per point, per-lane sorted top-16, LDS merge.
__global__ __launch_bounds__(128) void knn_kernel(const float* __restrict__ xyz,
                                                  int* __restrict__ knn_idx, int N) {
  __shared__ float sd[4][512];
  __shared__ int   si[4][512];
  int gwave = (blockIdx.x * blockDim.x + threadIdx.x) >> 5;
  int lane  = threadIdx.x & 31;
  int w     = threadIdx.x >> 5;
  int b = gwave / N, n = gwave % N;
  const float* base = xyz + (size_t)b * N * 3;
  float px = base[n * 3 + 0], py = base[n * 3 + 1], pz = base[n * 3 + 2];

  float bd[KNN]; int bi[KNN];
#pragma unroll
  for (int i = 0; i < KNN; ++i) { bd[i] = 3.4e38f; bi[i] = 0x7fffffff; }

  for (int m = lane; m < N; m += 32) {
    float dx = base[m * 3 + 0] - px;
    float dy = base[m * 3 + 1] - py;
    float dz = base[m * 3 + 2] - pz;
    float d = dx * dx + dy * dy + dz * dz;
    if (d < bd[KNN - 1] || (d == bd[KNN - 1] && m < bi[KNN - 1])) {
      int j = KNN - 1;
      while (j > 0 && (bd[j - 1] > d || (bd[j - 1] == d && bi[j - 1] > m))) {
        bd[j] = bd[j - 1]; bi[j] = bi[j - 1]; --j;
      }
      bd[j] = d; bi[j] = m;
    }
  }
#pragma unroll
  for (int i = 0; i < KNN; ++i) { sd[w][lane * KNN + i] = bd[i]; si[w][lane * KNN + i] = bi[i]; }
  __syncthreads();
  if (lane == 0) {
    int* out = knn_idx + (size_t)(b * N + n) * KNN;
    for (int kk = 0; kk < KNN; ++kk) {
      float best = 3.4e38f; int bestI = 0x7fffffff, bestJ = 0;
      for (int j = 0; j < 512; ++j) {
        float dj = sd[w][j]; int ij = si[w][j];
        if (dj < best || (dj == best && ij < bestI)) { best = dj; bestI = ij; bestJ = j; }
      }
      sd[w][bestJ] = 3.4e38f;
      out[kk] = bestI;
    }
  }
}

// ---------------------------------------------------------------------------
// f16 WMMA GEMM: C[M,N] = A[M,K] * Bpacked[K,N] + bias (f32 out).
// One wave per 16x64 strip: 4 accumulators share each A fragment.
__global__ __launch_bounds__(256) void gemm_f16_wmma(const _Float16* __restrict__ A,
                                                     const v16h* __restrict__ BP,
                                                     const float* __restrict__ bias,
                                                     float* __restrict__ C,
                                                     int M, int N, int K) {
  int wave = (int)((blockIdx.x * blockDim.x + threadIdx.x) >> 5);
  int lane = threadIdx.x & 31;
  int ngroups = N >> 6;                 // groups of 4 N-tiles
  int ntiles  = N >> 4;
  int tm = wave / ngroups;
  int tg = wave % ngroups;
  if (tm * 16 >= M) return;             // wave-uniform
  int tn0 = tg * 4;

  v8f acc0 = {}, acc1 = {}, acc2 = {}, acc3 = {};
  const _Float16* Abase = A + (size_t)tm * 16 * K;
  for (int k = 0; k < K; k += 32) {
    if (k + 32 < K) __builtin_prefetch(Abase + (lane & 15) * K + k + 32, 0, 0);
    v16h af = load_a_frag(Abase + k, K, lane);
    const v16h* bp = BP + ((size_t)(k >> 5) * ntiles + tn0) * 32 + lane;
    v16h b0 = bp[0];
    v16h b1 = bp[32];
    v16h b2 = bp[64];
    v16h b3 = bp[96];
    acc0 = __builtin_amdgcn_wmma_f32_16x16x32_f16(false, af, false, b0, (short)0, acc0, false, false);
    acc1 = __builtin_amdgcn_wmma_f32_16x16x32_f16(false, af, false, b1, (short)0, acc1, false, false);
    acc2 = __builtin_amdgcn_wmma_f32_16x16x32_f16(false, af, false, b2, (short)0, acc2, false, false);
    acc3 = __builtin_amdgcn_wmma_f32_16x16x32_f16(false, af, false, b3, (short)0, acc3, false, false);
  }
  int hlf = (lane >> 4) & 1;
  int nc  = lane & 15;
  v8f accs[4] = {acc0, acc1, acc2, acc3};
#pragma unroll
  for (int i = 0; i < 4; ++i) {
    int n = (tn0 + i) * 16 + nc;
    float bv = bias ? bias[n] : 0.0f;
#pragma unroll
    for (int r = 0; r < 8; ++r)
      C[(size_t)(tm * 16 + r + 8 * hlf) * N + n] = accs[i][r] + bv;
  }
}

// ---------------------------------------------------------------------------
// Wave-per-row layernorm. Optional residual add, f32 and/or f16 outputs.
__global__ __launch_bounds__(256) void layernorm_kernel(const float* __restrict__ X,
                                                        const float* __restrict__ g,
                                                        const float* __restrict__ beta,
                                                        const float* __restrict__ residual,
                                                        float* __restrict__ Yf,
                                                        _Float16* __restrict__ Yh,
                                                        int M, int D) {
  int row  = (int)((blockIdx.x * blockDim.x + threadIdx.x) >> 5);
  int lane = threadIdx.x & 31;
  if (row >= M) return;
  const float* x = X + (size_t)row * D;
  float s = 0.f, s2 = 0.f;
  for (int d = lane; d < D; d += 32) { float v = x[d]; s += v; s2 += v * v; }
#pragma unroll
  for (int off = 16; off; off >>= 1) { s += __shfl_xor(s, off, 32); s2 += __shfl_xor(s2, off, 32); }
  float mean = s / (float)D;
  float var  = s2 / (float)D - mean * mean;
  float inv  = rsqrtf(var + 1e-5f);
  for (int d = lane; d < D; d += 32) {
    float y = (x[d] - mean) * inv * g[d] + beta[d];
    if (residual) y += residual[(size_t)row * D + d];
    if (Yf) Yf[(size_t)row * D + d] = y;
    if (Yh) Yh[(size_t)row * D + d] = (_Float16)y;
  }
}

// ---------------------------------------------------------------------------
// Fused per-point attention: pos_enc (16x512x512 WMMA GEMM, A in LDS,
// packed d2_w from L2), logits, softmax, weighted-V.
// One 256-thread block (8 waves) per point; each wave owns 4 N-tiles.
__global__ __launch_bounds__(256) void attn_kernel(const float* __restrict__ xyz,
                                                   const int* __restrict__ knn_idx,
                                                   const float* __restrict__ d1_w,
                                                   const float* __restrict__ d1_b,
                                                   const v16h* __restrict__ d2_p,
                                                   const float* __restrict__ d2_b,
                                                   const float* __restrict__ q,
                                                   const float* __restrict__ xk,
                                                   const float* __restrict__ xv,
                                                   float* __restrict__ res,
                                                   float* __restrict__ attn_out,
                                                   int N) {
  __shared__ __align__(16) _Float16 s_hid[KNN * DMODEL];   // 16 KB
  __shared__ float    s_pos[KNN * DMODEL];                 // 32 KB
  __shared__ float    s_rel[KNN * 3];
  __shared__ float    s_att[KNN];
  __shared__ int      s_idx[KNN];

  int pt  = blockIdx.x;          // b*N + n
  int b   = pt / N;
  int n   = pt % N;
  int tid = threadIdx.x;
  int wv  = tid >> 5;
  int lane = tid & 31;

  const int* idxg = knn_idx + (size_t)pt * KNN;
  if (tid < KNN) s_idx[tid] = idxg[tid];
  __syncthreads();
  const float* xb = xyz + (size_t)b * N * 3;
  if (tid < KNN * 3) {
    int j = tid / 3, c = tid % 3;
    s_rel[tid] = xb[n * 3 + c] - xb[(size_t)s_idx[j] * 3 + c];
  }
  __syncthreads();

  // hidden = relu(rel @ d1_w + d1_b) : [16,512] -> LDS as f16
  for (int e = tid; e < KNN * DMODEL; e += 256) {
    int j = e / DMODEL, d = e % DMODEL;
    float h = d1_b[d] + s_rel[j * 3 + 0] * d1_w[0 * DMODEL + d]
                      + s_rel[j * 3 + 1] * d1_w[1 * DMODEL + d]
                      + s_rel[j * 3 + 2] * d1_w[2 * DMODEL + d];
    s_hid[e] = (_Float16)(h > 0.f ? h : 0.f);
  }
  __syncthreads();

  // pos = hidden @ d2_w + d2_b : M=16, K=512, N=512; 32 tiles, 4 per wave.
  {
    const int ntiles = DMODEL / 16;   // 32
    int t0 = wv * 4;
    v8f acc0 = {}, acc1 = {}, acc2 = {}, acc3 = {};
    for (int k = 0; k < DMODEL; k += 32) {
      v16h af = load_a_frag(s_hid + k, DMODEL, lane);
      const v16h* bp = d2_p + ((size_t)(k >> 5) * ntiles + t0) * 32 + lane;
      v16h b0 = bp[0];
      v16h b1 = bp[32];
      v16h b2 = bp[64];
      v16h b3 = bp[96];
      acc0 = __builtin_amdgcn_wmma_f32_16x16x32_f16(false, af, false, b0, (short)0, acc0, false, false);
      acc1 = __builtin_amdgcn_wmma_f32_16x16x32_f16(false, af, false, b1, (short)0, acc1, false, false);
      acc2 = __builtin_amdgcn_wmma_f32_16x16x32_f16(false, af, false, b2, (short)0, acc2, false, false);
      acc3 = __builtin_amdgcn_wmma_f32_16x16x32_f16(false, af, false, b3, (short)0, acc3, false, false);
    }
    int hlf = (lane >> 4) & 1;
    int nc  = lane & 15;
    v8f accs[4] = {acc0, acc1, acc2, acc3};
#pragma unroll
    for (int i = 0; i < 4; ++i) {
      int nn = (t0 + i) * 16 + nc;
      float bv = d2_b[nn];
#pragma unroll
      for (int r = 0; r < 8; ++r)
        s_pos[(size_t)(r + 8 * hlf) * DMODEL + nn] = accs[i][r] + bv;
    }
  }
  __syncthreads();

  // logits[j] = q . (k_gathered[j] + pos[j]) / sqrt(DM)   (2 rows per wave)
  const float scale = 0.044194173824159216f;  // 1/sqrt(512)
  const float* qn = q + (size_t)pt * DMODEL;
#pragma unroll
  for (int rj = 0; rj < 2; ++rj) {
    int j = wv * 2 + rj;
    const float* krow = xk + (size_t)(b * N + s_idx[j]) * DMODEL;
    float s = 0.f;
    for (int d = lane; d < DMODEL; d += 32) s += qn[d] * (krow[d] + s_pos[j * DMODEL + d]);
#pragma unroll
    for (int off = 16; off; off >>= 1) s += __shfl_xor(s, off, 32);
    if (lane == 0) s_att[j] = s * scale;
  }
  __syncthreads();

  // softmax over K=16 (wave 0)
  if (tid < 32) {
    float l = (lane < KNN) ? s_att[lane] : -3.4e38f;
    float mx = l;
#pragma unroll
    for (int off = 16; off; off >>= 1) mx = fmaxf(mx, __shfl_xor(mx, off, 32));
    float e = (lane < KNN) ? __expf(l - mx) : 0.f;
    float se = e;
#pragma unroll
    for (int off = 16; off; off >>= 1) se += __shfl_xor(se, off, 32);
    float a = e / se;
    if (lane < KNN) {
      s_att[lane] = a;
      attn_out[(size_t)pt * KNN + lane] = a;
    }
  }
  __syncthreads();

  // res[d] = sum_j attn[j] * (v_gathered[j][d] + pos[j][d])
  for (int d = tid; d < DMODEL; d += 256) {
    float acc = 0.f;
#pragma unroll
    for (int j = 0; j < KNN; ++j) {
      const float* vrow = xv + (size_t)(b * N + s_idx[j]) * DMODEL;
      acc += s_att[j] * (vrow[d] + s_pos[j * DMODEL + d]);
    }
    res[(size_t)pt * DMODEL + d] = acc;
  }
}

// ---------------------------------------------------------------------------
static char* carve(char*& p, size_t bytes) {
  char* r = p;
  p += (bytes + 255) & ~(size_t)255;
  return r;
}

extern "C" void kernel_launch(void* const* d_in, const int* in_sizes, int n_in,
                              void* d_out, int out_size, void* d_ws, size_t ws_size,
                              hipStream_t stream) {
  (void)in_sizes; (void)n_in; (void)out_size; (void)ws_size;
  const float* xyz   = (const float*)d_in[0];
  const float* feats = (const float*)d_in[1];
  const float* fc1_w = (const float*)d_in[2];
  const float* fc1_b = (const float*)d_in[3];
  const float* ln1_g = (const float*)d_in[4];
  const float* ln1_b = (const float*)d_in[5];
  const float* fc2_w = (const float*)d_in[6];
  const float* fc2_b = (const float*)d_in[7];
  const float* ln2_g = (const float*)d_in[8];
  const float* ln2_b = (const float*)d_in[9];
  const float* d1_w  = (const float*)d_in[10];
  const float* d1_b  = (const float*)d_in[11];
  const float* d2_w  = (const float*)d_in[12];
  const float* d2_b  = (const float*)d_in[13];
  const float* wq    = (const float*)d_in[14];
  const float* wk    = (const float*)d_in[15];
  const float* wv    = (const float*)d_in[16];

  const int M = BDIM * NPTS;           // 16384 rows

  char* p = (char*)d_ws;
  int*       knn   = (int*)      carve(p, (size_t)M * KNN * 4);
  _Float16*  featH = (_Float16*) carve(p, (size_t)M * DP * 2);
  v16h*      fc1wP = (v16h*)     carve(p, (size_t)DP * DMODEL * 2);
  v16h*      wqP   = (v16h*)     carve(p, (size_t)DMODEL * DMODEL * 2);
  v16h*      wkP   = (v16h*)     carve(p, (size_t)DMODEL * DMODEL * 2);
  v16h*      wvP   = (v16h*)     carve(p, (size_t)DMODEL * DMODEL * 2);
  v16h*      d2wP  = (v16h*)     carve(p, (size_t)DMODEL * DMODEL * 2);
  v16h*      fc2wP = (v16h*)     carve(p, (size_t)DMODEL * DP * 2);
  float*     x1    = (float*)    carve(p, (size_t)M * DMODEL * 4);
  _Float16*  xH    = (_Float16*) carve(p, (size_t)M * DMODEL * 2);
  float*     qbuf  = (float*)    carve(p, (size_t)M * DMODEL * 4);
  float*     xkbuf = (float*)    carve(p, (size_t)M * DMODEL * 4);
  float*     xvbuf = (float*)    carve(p, (size_t)M * DMODEL * 4);
  float*     resb  = (float*)    carve(p, (size_t)M * DMODEL * 4);
  _Float16*  resH  = (_Float16*) carve(p, (size_t)M * DMODEL * 2);
  float*     y2    = (float*)    carve(p, (size_t)M * DP * 4);

  float* out_res  = (float*)d_out;                 // [B,N,DP]
  float* out_attn = out_res + (size_t)M * DP;      // [B,N,1,K]

  auto cvt = [&](const float* src, _Float16* dst, int n) {
    convert_f16_kernel<<<(n + 255) / 256, 256, 0, stream>>>(src, dst, n);
  };
  auto repack = [&](const float* src, v16h* dst, int K, int N) {
    int total = (K / 32) * (N / 16) * 32;
    repack_b_kernel<<<(total + 255) / 256, 256, 0, stream>>>(src, dst, K, N);
  };
  auto gemm = [&](const _Float16* A, const v16h* BP, const float* bias,
                  float* C, int Mm, int Nm, int Kk) {
    int waves = (Mm / 16) * (Nm / 64);
    gemm_f16_wmma<<<(waves * 32) / 256, 256, 0, stream>>>(A, BP, bias, C, Mm, Nm, Kk);
  };

  // weight repack (f32 -> f16 fragment-major) + activation convert
  cvt(feats, featH, M * DP);
  repack(fc1_w, fc1wP, DP, DMODEL);
  repack(wq, wqP, DMODEL, DMODEL);
  repack(wk, wkP, DMODEL, DMODEL);
  repack(wv, wvP, DMODEL, DMODEL);
  repack(d2_w, d2wP, DMODEL, DMODEL);
  repack(fc2_w, fc2wP, DMODEL, DP);

  // KNN (wave per point; 4 waves per block)
  knn_kernel<<<M / 4, 128, 0, stream>>>(xyz, knn, NPTS);

  // x = layernorm(features @ fc1_w + fc1_b)
  gemm(featH, fc1wP, fc1_b, x1, M, DMODEL, DP);
  layernorm_kernel<<<(M + 7) / 8, 256, 0, stream>>>(x1, ln1_g, ln1_b, nullptr,
                                                    nullptr, xH, M, DMODEL);

  // q / k / v projections
  gemm(xH, wqP, nullptr, qbuf,  M, DMODEL, DMODEL);
  gemm(xH, wkP, nullptr, xkbuf, M, DMODEL, DMODEL);
  gemm(xH, wvP, nullptr, xvbuf, M, DMODEL, DMODEL);

  // fused pos_enc + attention, one block per point
  attn_kernel<<<M, 256, 0, stream>>>(xyz, knn, d1_w, d1_b, d2wP, d2_b,
                                     qbuf, xkbuf, xvbuf, resb, out_attn, NPTS);

  // out = layernorm(res @ fc2_w + fc2_b) + features
  cvt(resb, resH, M * DMODEL);
  gemm(resH, fc2wP, fc2_b, y2, M, DP, DMODEL);
  layernorm_kernel<<<(M + 7) / 8, 256, 0, stream>>>(y2, ln2_g, ln2_b, feats,
                                                    out_res, nullptr, M, DP);
}
